// TopologicalMap_15418932592734
// MI455X (gfx1250) — matrix operand order repaired and checked by reading, not compiled, not run on hardware
//
#include <hip/hip_runtime.h>

// SOM squared-distance: out[b][n] = ||x_b||^2 + ||w_:,n||^2 - 2 * x_b . w_:,n
// x: (8192, 64) f32   w: (64, 16384) f32   out: (8192, 16384) f32
//
// Strategy (MI455X / gfx1250):
//  - Output stream (512 MiB) dominates -> fuse epilogue, non-temporal stores.
//  - GEMM via split-bf16 (hi/lo) with V_WMMA_F32_16X16X32_BF16:
//      cross ~= xhi.whi + xhi.wlo + xlo.whi  (f32-grade accuracy, 8x FLOP/instr
//      vs the f32 16x16x4 WMMA, so 3 products still ~2.7x fewer matrix instrs).
//  - 128x128 output tile per 256-thread (8-wave) workgroup; LDS-staged tiles.

typedef __attribute__((ext_vector_type(16))) __bf16 v16bf;
typedef __attribute__((ext_vector_type(8)))  float  v8f;

#define SOM_B 8192
#define SOM_D 64
#define SOM_N 16384
#define TM 128
#define TN 128
#define LDP 72   // padded LDS row stride in bf16 elements (144 B, 16B-aligned)

union Frag16 { v16bf v; uint4 q[2]; };
union BF4    { __bf16 b[4]; uint2 u2; };

__device__ __forceinline__ void bsplit(float f, __bf16& hi, __bf16& lo) {
    hi = (__bf16)f;
    lo = (__bf16)(f - (float)hi);
}

__global__ __launch_bounds__(256, 2)
void som_dist_wmma(const float* __restrict__ x,
                   const float* __restrict__ w,
                   float* __restrict__ out)
{
    __shared__ __align__(16) __bf16 sXhi[TM][LDP];
    __shared__ __align__(16) __bf16 sXlo[TM][LDP];
    __shared__ __align__(16) __bf16 sWhi[TN][LDP];   // transposed: [n][k]
    __shared__ __align__(16) __bf16 sWlo[TN][LDP];
    __shared__ float sX2[TM];
    __shared__ float sW2[TN];

    const int tid     = threadIdx.x;
    const int colBase = blockIdx.x * TN;
    const int rowBase = blockIdx.y * TM;

    // ---- stage x tile: (TM x 64) f32 -> hi/lo bf16, row-major ----
    for (int i = tid; i < (TM * SOM_D) / 4; i += 256) {
        const int r = i >> 4;             // (i*4)/64
        const int c = (i & 15) << 2;      // (i*4)%64
        const float4 v = *(const float4*)(x + (size_t)(rowBase + r) * SOM_D + c);
        BF4 hi, lo;
        bsplit(v.x, hi.b[0], lo.b[0]);
        bsplit(v.y, hi.b[1], lo.b[1]);
        bsplit(v.z, hi.b[2], lo.b[2]);
        bsplit(v.w, hi.b[3], lo.b[3]);
        *(uint2*)&sXhi[r][c] = hi.u2;
        *(uint2*)&sXlo[r][c] = lo.u2;
    }

    // ---- stage w tile: (64 x TN) f32 -> hi/lo bf16, TRANSPOSED [n][k] ----
    for (int i = tid; i < (SOM_D * TN) / 4; i += 256) {
        const int k = i >> 5;             // (i*4)/128
        const int j = (i & 31) << 2;      // (i*4)%128
        const float4 v = *(const float4*)(w + (size_t)k * SOM_N + colBase + j);
        const float f[4] = {v.x, v.y, v.z, v.w};
        #pragma unroll
        for (int e = 0; e < 4; ++e) {
            __bf16 h, l;
            bsplit(f[e], h, l);
            sWhi[j + e][k] = h;
            sWlo[j + e][k] = l;
        }
    }
    __syncthreads();

    // ---- squared norms from the staged (hi+lo) values ----
    if (tid < TM) {
        float s = 0.f;
        #pragma unroll 8
        for (int k = 0; k < SOM_D; ++k) {
            const float f = (float)sXhi[tid][k] + (float)sXlo[tid][k];
            s += f * f;
        }
        sX2[tid] = s;
    } else {
        const int n = tid - TM;
        float s = 0.f;
        #pragma unroll 8
        for (int k = 0; k < SOM_D; ++k) {
            const float f = (float)sWhi[n][k] + (float)sWlo[n][k];
            s += f * f;
        }
        sW2[n] = s;
    }
    __syncthreads();

    // ---- WMMA compute: each wave owns a 16-row strip x 128 cols ----
    const int wave  = tid >> 5;
    const int lane  = tid & 31;
    const int laneN = lane & 15;
    const int half  = lane >> 4;      // 0: lanes 0-15, 1: lanes 16-31
    const int r0    = wave * 16;

    v8f acc[8] = {};

    #pragma unroll
    for (int kb = 0; kb < SOM_D; kb += 32) {
        // A fragment (16x32 bf16): lanes 0-15 hold K {kb..kb+7, kb+16..kb+23},
        // lanes 16-31 hold K {kb+8..kb+15, kb+24..kb+31}; M = laneN.
        Frag16 ahi, alo;
        const int aRow = r0 + laneN;
        const int aK   = kb + half * 8;
        ahi.q[0] = *(const uint4*)&sXhi[aRow][aK];
        ahi.q[1] = *(const uint4*)&sXhi[aRow][aK + 16];
        alo.q[0] = *(const uint4*)&sXlo[aRow][aK];
        alo.q[1] = *(const uint4*)&sXlo[aRow][aK + 16];

        #pragma unroll
        for (int t = 0; t < 8; ++t) {
            // B fragment (32x16 bf16): N = laneN, lanes 0-15 hold K {kb..kb+7, kb+16..kb+23},
            // lanes 16-31 hold K {kb+8..kb+15, kb+24..kb+31} (matches A).
            Frag16 bhi, blo;
            const int bRow = t * 16 + laneN;
            const int bK   = kb + half * 8;
            bhi.q[0] = *(const uint4*)&sWhi[bRow][bK];
            bhi.q[1] = *(const uint4*)&sWhi[bRow][bK + 16];
            blo.q[0] = *(const uint4*)&sWlo[bRow][bK];
            blo.q[1] = *(const uint4*)&sWlo[bRow][bK + 16];

            acc[t] = __builtin_amdgcn_wmma_f32_16x16x32_bf16(
                false, ahi.v, false, bhi.v, (short)0, acc[t], false, false);
            acc[t] = __builtin_amdgcn_wmma_f32_16x16x32_bf16(
                false, ahi.v, false, blo.v, (short)0, acc[t], false, false);
            acc[t] = __builtin_amdgcn_wmma_f32_16x16x32_bf16(
                false, alo.v, false, bhi.v, (short)0, acc[t], false, false);
        }
    }

    // ---- fused epilogue: x2 + w2 - 2*cross, streamed with NT stores ----
    // C/D layout: VGPR v -> M = v + 8*half ; N = laneN.
    #pragma unroll
    for (int t = 0; t < 8; ++t) {
        const int   col = colBase + t * 16 + laneN;
        const float w2v = sW2[t * 16 + laneN];
        #pragma unroll
        for (int v = 0; v < 8; ++v) {
            const int   m   = r0 + v + half * 8;
            const float val = sX2[m] + w2v - 2.0f * acc[t][v];
            __builtin_nontemporal_store(val,
                out + (size_t)(rowBase + m) * SOM_N + col);
        }
    }
}

extern "C" void kernel_launch(void* const* d_in, const int* in_sizes, int n_in,
                              void* d_out, int out_size, void* d_ws, size_t ws_size,
                              hipStream_t stream) {
    const float* x = (const float*)d_in[0];   // (8192, 64)
    const float* w = (const float*)d_in[1];   // (64, 16384)
    float* out     = (float*)d_out;           // (8192, 16384)

    dim3 grid(SOM_N / TN, SOM_B / TM);        // (128, 64)
    dim3 block(256);
    som_dist_wmma<<<grid, block, 0, stream>>>(x, w, out);
}
